// CausalSelfAttention_25451976196158
// MI455X (gfx1250) — matrix operand order repaired and checked by reading, not compiled, not run on hardware
//
#include <hip/hip_runtime.h>
#include <hip/hip_bf16.h>

// CDNA5 / gfx1250 causal self-attention forward.
// GEMMs: v_wmma_f32_16x16x32_bf16 with TDM (tensor_load_to_lds) double-buffered
// weight staging in LDS. Attention: flash-style, WMMA scores + WMMA P@V with an
// LDS restripe of P from C-layout to A-layout.
// B=8, T=1024, C=768, H=12, HD=64.

typedef __attribute__((ext_vector_type(16))) __bf16       bf16x16;
typedef __attribute__((ext_vector_type(8)))  float        f32x8;
typedef __attribute__((ext_vector_type(4)))  unsigned int u32x4;
typedef __attribute__((ext_vector_type(8)))  int          i32x8;
typedef __attribute__((ext_vector_type(4)))  int          i32x4;

#define B_  8
#define T_  1024
#define C_  768
#define H_  12
#define HD_ 64
#define M_  (B_ * T_)      // 8192
#define N3C (3 * C_)       // 2304

static __device__ __forceinline__ unsigned short f2bf(float f) {
    union { float f; unsigned int u; } v; v.f = f;
    unsigned int u = v.u;
    u += 0x7fffu + ((u >> 16) & 1u);     // round-to-nearest-even
    return (unsigned short)(u >> 16);
}

static __device__ __forceinline__ f32x8 wmma_bf16(bf16x16 a, bf16x16 b, f32x8 c) {
    return __builtin_amdgcn_wmma_f32_16x16x32_bf16(
        /*neg_a=*/false, a, /*neg_b=*/false, b,
        /*c_mod=*/(short)0, c, /*reuse_a=*/false, /*reuse_b=*/false);
}

// A fragment (16x32 bf16, row-major source, leading dim ldk elements).
// lane: M = lane&15 ; elems e -> K = k0 + 16*(lane>>4) + e (contiguous).
static __device__ __forceinline__ bf16x16
load_a_frag(const unsigned short* a, int ldk, int m0, int k0, int lane) {
    const int m  = m0 + (lane & 15);
    const int kk = k0 + ((lane >> 4) << 4);
    const unsigned short* p = a + (size_t)m * ldk + kk;
    union { u32x4 q[2]; bf16x16 v; } u;
    u.q[0] = *(const u32x4*)(p);
    u.q[1] = *(const u32x4*)(p + 8);
    return u.v;
}

// B fragment (32x16 bf16) from a K-major (transposed) source BT[N][K], ld ldk.
// lane: N = n0 + (lane&15) ; elems e -> K = k0 + 16*(lane>>4) + e (contiguous).
static __device__ __forceinline__ bf16x16
load_b_frag_T(const unsigned short* bt, int ldk, int n0, int k0, int lane) {
    const int n  = n0 + (lane & 15);
    const int kk = k0 + ((lane >> 4) << 4);
    const unsigned short* p = bt + (size_t)n * ldk + kk;
    union { u32x4 q[2]; bf16x16 v; } u;
    u.q[0] = *(const u32x4*)(p);
    u.q[1] = *(const u32x4*)(p + 8);
    return u.v;
}

// B fragment from an LDS-staged [64 x 32] bf16 row-major tile. f = 16-col subtile.
static __device__ __forceinline__ bf16x16
lds_b_frag(const unsigned short* base, int f, int lane) {
    const int row = f * 16 + (lane & 15);
    const int kk  = (lane >> 4) << 4;
    const unsigned short* p = base + row * 32 + kk;
    union { u32x4 q[2]; bf16x16 v; } u;
    u.q[0] = *(const u32x4*)(p);        // ds_load_b128
    u.q[1] = *(const u32x4*)(p + 8);
    return u.v;
}

// ---------------- conversion kernels ----------------

__global__ void cvt_f32_bf16_kernel(const float* __restrict__ in,
                                    unsigned short* __restrict__ out, int n) {
    for (int i = blockIdx.x * blockDim.x + threadIdx.x; i < n;
         i += gridDim.x * blockDim.x)
        out[i] = f2bf(in[i]);
}

// w[K][N] f32 (row-major) -> wT[N][K] bf16
__global__ void cvt_wT_kernel(const float* __restrict__ w,
                              unsigned short* __restrict__ wT, int K, int N) {
    const int total = K * N;
    for (int i = blockIdx.x * blockDim.x + threadIdx.x; i < total;
         i += gridDim.x * blockDim.x) {
        const int k = i / N, n = i - k * N;
        wT[(size_t)n * K + k] = f2bf(w[i]);
    }
}

// ---------------- shared GEMM mainloop with TDM-staged B tiles ----------------
// Block = 8 waves sharing one 64-column strip of wT. Per K-step of 32, wave 0
// DMAs the next 64x32 bf16 B tile into the alternate LDS buffer with the Tensor
// Data Mover while all waves compute on the current one. A fragments are
// register-prefetched one step ahead.

static __device__ __forceinline__ void tdm_issue_b_tile(
    const unsigned short* __restrict__ wT, int ndim1, int n0, int k0,
    unsigned lds_addr)
{
    // D# per CDNA5 ISA: 2-D tile, dim0 = K (contiguous).
    const unsigned long long ga =
        (unsigned long long)(size_t)(wT + (size_t)n0 * C_ + k0);
    u32x4 g0;
    g0.x = 1u;                                            // count=1 (valid D#)
    g0.y = lds_addr;                                      // lds_addr (bytes)
    g0.z = (unsigned)(ga & 0xffffffffull);                // global_addr[31:0]
    g0.w = (unsigned)((ga >> 32) & 0x01ffffffull)         // global_addr[56:32]
           | 0x80000000u;                                 // type=2 ("image")
    i32x8 g1;
    g1[0] = 0x00010000;                                   // data_size=1 -> 2B/elem
    g1[1] = (int)((unsigned)C_ << 16);                    // tensor_dim0 = 768 (K)
    g1[2] = (int)((unsigned)ndim1 << 16);                 // tensor_dim1 (N lines)
    g1[3] = (int)(32u << 16);                             // tile_dim0 = 32
    g1[4] = 64;                                           // tile_dim1 = 64
    g1[5] = C_;                                           // tensor_dim0_stride
    g1[6] = 0;
    g1[7] = 0;
    const i32x4 z4 = {0, 0, 0, 0};
#if __clang_major__ >= 23
    const i32x8 z8 = {0, 0, 0, 0, 0, 0, 0, 0};
    __builtin_amdgcn_tensor_load_to_lds(g0, g1, z4, z4, z8, 0);
#else
    __builtin_amdgcn_tensor_load_to_lds(g0, g1, z4, z4, 0);
#endif
}

static __device__ __forceinline__ void gemm_mainloop(
    const unsigned short* __restrict__ aglb, const unsigned short* __restrict__ wT,
    unsigned short* s0, unsigned short* s1, int ndim1,
    int m0, int n0, int lane, int wv,
    f32x8& c0, f32x8& c1, f32x8& c2, f32x8& c3)
{
    const unsigned lds0 = (unsigned)(size_t)s0;   // low 32 bits = LDS byte offset
    const unsigned lds1 = (unsigned)(size_t)s1;

    if (wv == 0) tdm_issue_b_tile(wT, ndim1, n0, 0, lds0);
    bf16x16 a_cur = load_a_frag(aglb, C_, m0, 0, lane);
    if (wv == 0) __builtin_amdgcn_s_wait_tensorcnt(0);
    __syncthreads();

    for (int k0 = 0; k0 < C_; k0 += 32) {
        const int buf  = (k0 >> 5) & 1;
        const bool more = (k0 + 32) < C_;
        if (wv == 0 && more)
            tdm_issue_b_tile(wT, ndim1, n0, k0 + 32, buf ? lds0 : lds1);
        const bf16x16 a_next =
            load_a_frag(aglb, C_, m0, more ? (k0 + 32) : k0, lane);

        const unsigned short* bp = buf ? s1 : s0;
        c0 = wmma_bf16(a_cur, lds_b_frag(bp, 0, lane), c0);
        c1 = wmma_bf16(a_cur, lds_b_frag(bp, 1, lane), c1);
        c2 = wmma_bf16(a_cur, lds_b_frag(bp, 2, lane), c2);
        c3 = wmma_bf16(a_cur, lds_b_frag(bp, 3, lane), c3);

        if (wv == 0 && more) __builtin_amdgcn_s_wait_tensorcnt(0);
        __syncthreads();                 // publish next buffer / retire this one
        a_cur = a_next;
    }
}

// ---------------- QKV projection GEMM ----------------
// qkv = x @ W_attn + b ; scatter into q[BH,T,64], k[BH,T,64], vT[BH,64,T] (bf16)

__global__ __launch_bounds__(256) void qkv_gemm_kernel(
    const unsigned short* __restrict__ xb, const unsigned short* __restrict__ wT,
    const float* __restrict__ bias, unsigned short* __restrict__ qg,
    unsigned short* __restrict__ kg, unsigned short* __restrict__ vTg)
{
    __shared__ __align__(16) unsigned short bstage[2][64 * 32];  // 2 x 4KB
    const int lane = threadIdx.x & 31;
    const int wv   = threadIdx.x >> 5;
    const int nstrip = blockIdx.x % (N3C / 64);   // 36 strips
    const int mblk   = blockIdx.x / (N3C / 64);   // 64 row-blocks of 128
    const int n0 = nstrip * 64;
    const int m0 = (mblk * 8 + wv) * 16;

    f32x8 a0 = {}, a1 = {}, a2 = {}, a3 = {};
    gemm_mainloop(xb, wT, &bstage[0][0], &bstage[1][0], N3C,
                  m0, n0, lane, wv, a0, a1, a2, a3);

    const int half = lane >> 4, nl = lane & 15;
    auto epi = [&](const f32x8& acc, int f) {
        const int col = n0 + f * 16 + nl;
        const float bv = bias[col];
        const int which = col / C_;
        const int c = col - which * C_;
        const int h = c >> 6, d = c & 63;
        #pragma unroll
        for (int r = 0; r < 8; ++r) {
            const int row = m0 + r + 8 * half;          // global token (b*T + t)
            const int b = row >> 10, t = row & (T_ - 1);
            const unsigned short v16 = f2bf(acc[r] + bv);
            const size_t bh = (size_t)(b * H_ + h);
            if (which == 0)      qg[(bh * T_ + t) * HD_ + d] = v16;
            else if (which == 1) kg[(bh * T_ + t) * HD_ + d] = v16;
            else                 vTg[(bh * HD_ + d) * T_ + t] = v16;
        }
    };
    epi(a0, 0); epi(a1, 1); epi(a2, 2); epi(a3, 3);
}

// ---------------- flash attention (one wave = one 16-row query tile) ----------------

__global__ __launch_bounds__(256) void attn_kernel(
    const unsigned short* __restrict__ qg, const unsigned short* __restrict__ kg,
    const unsigned short* __restrict__ vTg, unsigned short* __restrict__ yg)
{
    __shared__ __align__(16) unsigned short lds_p[8][16 * 32];  // per-wave P tile

    const int lane = threadIdx.x & 31;
    const int wv   = threadIdx.x >> 5;
    const int wid  = blockIdx.x * 8 + wv;              // 96 * 64 = 6144 waves exactly
    const int bh   = wid >> 6;
    const int qbase = (wid & 63) << 4;

    const unsigned short* qp = qg + (size_t)bh * (T_ * HD_);
    const unsigned short* kp = kg + (size_t)bh * (T_ * HD_);
    const unsigned short* vp = vTg + (size_t)bh * (HD_ * T_);
    const int half = lane >> 4, nl = lane & 15;

    const bf16x16 qa0 = load_a_frag(qp, HD_, qbase, 0,  lane);
    const bf16x16 qa1 = load_a_frag(qp, HD_, qbase, 32, lane);

    f32x8 o0 = {}, o1 = {}, o2 = {}, o3 = {};
    float mrow[8], lrow[8];
    #pragma unroll
    for (int r = 0; r < 8; ++r) { mrow[r] = -1e30f; lrow[r] = 0.f; }

    for (int j0 = 0; j0 < qbase + 16; j0 += 32) {
        // S = q @ k^T for 16x32 key tile (4 WMMAs, K=64)
        f32x8 s0 = {}, s1 = {};
        bf16x16 kb;
        kb = load_b_frag_T(kp, HD_, j0,      0,  lane); s0 = wmma_bf16(qa0, kb, s0);
        kb = load_b_frag_T(kp, HD_, j0,      32, lane); s0 = wmma_bf16(qa1, kb, s0);
        kb = load_b_frag_T(kp, HD_, j0 + 16, 0,  lane); s1 = wmma_bf16(qa0, kb, s1);
        kb = load_b_frag_T(kp, HD_, j0 + 16, 32, lane); s1 = wmma_bf16(qa1, kb, s1);

        // Issue the V B-fragment loads now so they overlap softmax + restripe.
        const bf16x16 vb0 = load_b_frag_T(vp, T_, 0,  j0, lane);
        const bf16x16 vb1 = load_b_frag_T(vp, T_, 16, j0, lane);
        const bf16x16 vb2 = load_b_frag_T(vp, T_, 32, j0, lane);
        const bf16x16 vb3 = load_b_frag_T(vp, T_, 48, j0, lane);

        // scale, causal mask, online softmax (rows r+8*half, cols across lanes 0..15)
        const int col0 = j0 + nl, col1 = j0 + 16 + nl;
        #pragma unroll
        for (int r = 0; r < 8; ++r) {
            const int row = qbase + r + 8 * half;
            float v0 = s0[r] * 0.125f; v0 = (col0 <= row) ? v0 : -1e30f;
            float v1 = s1[r] * 0.125f; v1 = (col1 <= row) ? v1 : -1e30f;
            float mx = fmaxf(v0, v1);
            mx = fmaxf(mx, __shfl_xor(mx, 1, 32));
            mx = fmaxf(mx, __shfl_xor(mx, 2, 32));
            mx = fmaxf(mx, __shfl_xor(mx, 4, 32));
            mx = fmaxf(mx, __shfl_xor(mx, 8, 32));
            const float mnew  = fmaxf(mrow[r], mx);
            const float alpha = __expf(mrow[r] - mnew);
            mrow[r] = mnew;
            const float p0 = __expf(v0 - mnew);
            const float p1 = __expf(v1 - mnew);
            s0[r] = p0; s1[r] = p1;
            float ps = p0 + p1;
            ps += __shfl_xor(ps, 1, 32);
            ps += __shfl_xor(ps, 2, 32);
            ps += __shfl_xor(ps, 4, 32);
            ps += __shfl_xor(ps, 8, 32);
            lrow[r] = lrow[r] * alpha + ps;
            o0[r] *= alpha; o1[r] *= alpha; o2[r] *= alpha; o3[r] *= alpha;
        }

        // C-layout -> A-layout restripe of P through LDS (bf16 16x32 row-major)
        #pragma unroll
        for (int r = 0; r < 8; ++r) {
            const int row = r + 8 * half;
            lds_p[wv][row * 32 + nl]      = f2bf(s0[r]);
            lds_p[wv][row * 32 + 16 + nl] = f2bf(s1[r]);
        }
        union { u32x4 qv[2]; bf16x16 v; } up;
        const int base = nl * 32 + half * 16;
        up.qv[0] = *(const u32x4*)&lds_p[wv][base];        // ds_load_b128
        up.qv[1] = *(const u32x4*)&lds_p[wv][base + 8];
        const bf16x16 pa = up.v;

        // O += P @ V
        o0 = wmma_bf16(pa, vb0, o0);
        o1 = wmma_bf16(pa, vb1, o1);
        o2 = wmma_bf16(pa, vb2, o2);
        o3 = wmma_bf16(pa, vb3, o3);
    }

    // normalize and store y as bf16 [B*T, C] for the projection GEMM
    const int b = bh / H_, h = bh - b * H_;
    #pragma unroll
    for (int r = 0; r < 8; ++r) {
        const float inv = 1.0f / lrow[r];
        const int t = qbase + r + 8 * half;
        unsigned short* dst = yg + ((size_t)(b * T_ + t)) * C_ + h * HD_ + nl;
        dst[0]  = f2bf(o0[r] * inv);
        dst[16] = f2bf(o1[r] * inv);
        dst[32] = f2bf(o2[r] * inv);
        dst[48] = f2bf(o3[r] * inv);
    }
}

// ---------------- output projection GEMM (f32 out) ----------------

__global__ __launch_bounds__(256) void proj_gemm_kernel(
    const unsigned short* __restrict__ yb, const unsigned short* __restrict__ wT,
    const float* __restrict__ bias, float* __restrict__ outp)
{
    __shared__ __align__(16) unsigned short bstage[2][64 * 32];
    const int lane = threadIdx.x & 31;
    const int wv   = threadIdx.x >> 5;
    const int nstrip = blockIdx.x % (C_ / 64);    // 12 strips
    const int mblk   = blockIdx.x / (C_ / 64);
    const int n0 = nstrip * 64;
    const int m0 = (mblk * 8 + wv) * 16;

    f32x8 a0 = {}, a1 = {}, a2 = {}, a3 = {};
    gemm_mainloop(yb, wT, &bstage[0][0], &bstage[1][0], C_,
                  m0, n0, lane, wv, a0, a1, a2, a3);

    const int half = lane >> 4, nl = lane & 15;
    auto epi = [&](const f32x8& acc, int f) {
        const int col = n0 + f * 16 + nl;
        const float bv = bias[col];
        #pragma unroll
        for (int r = 0; r < 8; ++r) {
            const int row = m0 + r + 8 * half;
            outp[(size_t)row * C_ + col] = acc[r] + bv;   // coalesced across lanes
        }
    };
    epi(a0, 0); epi(a1, 1); epi(a2, 2); epi(a3, 3);
}

// ---------------- launch ----------------

extern "C" void kernel_launch(void* const* d_in, const int* in_sizes, int n_in,
                              void* d_out, int out_size, void* d_ws, size_t ws_size,
                              hipStream_t stream) {
    (void)in_sizes; (void)n_in; (void)out_size; (void)ws_size;
    const float* x      = (const float*)d_in[0];
    const float* W_attn = (const float*)d_in[1];
    const float* b_attn = (const float*)d_in[2];
    const float* W_proj = (const float*)d_in[3];
    const float* b_proj = (const float*)d_in[4];
    float* out = (float*)d_out;

    // workspace carve-out (bf16 buffers), 256B aligned; total ~65 MB
    char* ws = (char*)d_ws;
    size_t off = 0;
    auto carve = [&](size_t elems) {
        size_t cur = off;
        off += (elems * 2 + 255) & ~(size_t)255;
        return (unsigned short*)(ws + cur);
    };
    unsigned short* x_bf    = carve((size_t)M_ * C_);
    unsigned short* wattnT  = carve((size_t)N3C * C_);
    unsigned short* wprojT  = carve((size_t)C_ * C_);
    unsigned short* q_bf    = carve((size_t)B_ * H_ * T_ * HD_);
    unsigned short* k_bf    = carve((size_t)B_ * H_ * T_ * HD_);
    unsigned short* vT_bf   = carve((size_t)B_ * H_ * T_ * HD_);
    unsigned short* y_bf    = carve((size_t)M_ * C_);

    // 1) precision conversions (+ weight transposes for contiguous B-fragments)
    cvt_f32_bf16_kernel<<<4096, 256, 0, stream>>>(x, x_bf, M_ * C_);
    cvt_wT_kernel<<<4096, 256, 0, stream>>>(W_attn, wattnT, C_, N3C);
    cvt_wT_kernel<<<2048, 256, 0, stream>>>(W_proj, wprojT, C_, C_);

    // 2) QKV GEMM: 64 row-blocks x 36 column strips, 8 waves/block
    qkv_gemm_kernel<<<64 * 36, 256, 0, stream>>>(x_bf, wattnT, b_attn,
                                                 q_bf, k_bf, vT_bf);

    // 3) flash attention: 96 heads * 64 query tiles = 6144 waves
    attn_kernel<<<6144 / 8, 256, 0, stream>>>(q_bf, k_bf, vT_bf, y_bf);

    // 4) output projection: 64 row-blocks x 12 column strips
    proj_gemm_kernel<<<64 * 12, 256, 0, stream>>>(y_bf, wprojT, b_proj, out);
}
